// CircuitGnn_27943057228426
// MI455X (gfx1250) — matrix operand order repaired and checked by reading, not compiled, not run on hardware
//
#include <hip/hip_runtime.h>
#include <hip/hip_bf16.h>

#define N_NODES 20000
#define N_EDGES 160000
#define ETOT    (N_EDGES + N_NODES)
#define NBATCH  40
#define HEADS   4
#define CCH     64
#define HC_MAX  256
#define SLOPE_ATT 0.2f
#define SLOPE_ACT 0.01f
#define LN_EPS 1e-5f

typedef __attribute__((ext_vector_type(16))) __bf16 v16bf;
typedef __attribute__((ext_vector_type(8)))  __bf16 v8bf;
typedef __attribute__((ext_vector_type(8)))  float  v8f;
typedef __attribute__((ext_vector_type(4)))  unsigned int v4u;
typedef __attribute__((ext_vector_type(8)))  int v8i;
typedef __attribute__((ext_vector_type(4)))  int v4i;

#if __has_builtin(__builtin_amdgcn_tensor_load_to_lds)
#define HAVE_TDM 1
#else
#define HAVE_TDM 0
#endif

// ---- ordered-int encoding for float atomicMax (monotone, handles negatives) ----
__device__ __forceinline__ int   enc_f(float f){ int i=__float_as_int(f); return (i>=0)? i : (i ^ 0x7fffffff); }
__device__ __forceinline__ float dec_f(int i){ int j=(i>=0)? i : (i ^ 0x7fffffff); return __int_as_float(j); }

// =====================================================================
// GEMM tiling constants
// =====================================================================
#define KT   64                 // K-slice per TDM transfer
#define TN   64                 // columns (of Y) per block
#define KROW 72                 // padded LDS row: 64 bf16 + 8 bf16 (16B) pad
                                // -> 144B rows, 36-word bank stride, conflict-free

// =====================================================================
// TDM: DMA a [TN rows x KT cols] bf16 tile of Wt[Nc][K] (row stride K) into
// LDS with per-row padding. D# per CDNA5 ISA 8.3/8.4:
//   group0 = {count=1, lds_addr, global_addr, type=2}
//   group1 = {data_size=2B, pad_enable, pad_interval=32dw, pad_amount=4dw,
//             tensor_dim0=K, tensor_dim1=Nc, tile=64x64, dim0_stride=K}
// =====================================================================
#if HAVE_TDM
__device__ __forceinline__ void tdm_load_tile(const __bf16* gsrc, unsigned lds_off,
                                              int K, int Nc)
{
    unsigned long long ga = (unsigned long long)(uintptr_t)gsrc;
    v4u g0;
    g0[0] = 1u;                                                // count=1, user mode
    g0[1] = lds_off;                                           // LDS byte offset
    g0[2] = (unsigned)(ga & 0xffffffffu);                      // global_addr[31:0]
    g0[3] = (unsigned)((ga >> 32) & 0x01ffffffu) | (2u << 30); // addr[56:32] | type=2
    v8i g1;
    g1[0] = (int)((1u << 16)      // data_size = 2 bytes (bf16)
                | (1u << 20)      // pad_enable
                | (4u << 22)      // pad_interval: 32 DWORDs (=128B row)
                | (3u << 25));    // pad_amount: 4 DWORDs (=16B)
    g1[1] = (int)(((unsigned)K & 0xffffu) << 16);              // tensor_dim0[15:0]
    g1[2] = (int)((((unsigned)K >> 16) & 0xffffu) | (((unsigned)Nc & 0xffffu) << 16));
    g1[3] = (int)((((unsigned)Nc >> 16) & 0xffffu) | ((unsigned)KT << 16)); // tile_dim0
    g1[4] = TN;                                                // tile_dim1, tile_dim2=0
    g1[5] = K;                                                 // tensor_dim0_stride[31:0]
    g1[6] = 0;
    g1[7] = 0;
    v4i z4 = {0, 0, 0, 0};
#if defined(__clang_major__) && (__clang_major__ >= 23)
    v8i z8 = {0, 0, 0, 0, 0, 0, 0, 0};
    __builtin_amdgcn_tensor_load_to_lds(g0, g1, z4, z4, z8, 0);
#else
    __builtin_amdgcn_tensor_load_to_lds(g0, g1, z4, z4, 0);
#endif
}
#endif

// =====================================================================
// WMMA GEMM: Y[M,Nc] = X[M,K] @ W[K,Nc] + bias
// X pre-converted to bf16 row-major; W pre-transposed to bf16 Wt[Nc][K].
// Block = 8 waves x 16 rows = 128-row band; each wave owns 4 n-tiles.
// Wt K-slices double-buffered in LDS via TDM (TENSORcnt pipelining);
// A = 2x global b128, B = 2x ds b128 per WMMA, zero in-loop converts.
// =====================================================================
__global__ void __launch_bounds__(256)
gemm_wmma(const __bf16* __restrict__ X, const __bf16* __restrict__ Wt,
          const float* __restrict__ bias, float* __restrict__ Y,
          int M, int K, int Nc)
{
    __shared__ __attribute__((aligned(16))) __bf16 lds_w[2][TN * KROW]; // 2 x 9 KB
    const int lane  = threadIdx.x & 31;
    const int wave  = threadIdx.x >> 5;
    const int nlane = lane & 15;
    const int half  = lane >> 4;
    const int mtile = blockIdx.x * 8 + wave;
    const int n0b   = blockIdx.y * TN;

    const int arow  = min(mtile * 16 + nlane, M - 1);  // clamp; garbage masked at store
    const __bf16* xrow = X + (size_t)arow * K;

    v8f acc[4];
    #pragma unroll
    for (int nt = 0; nt < 4; ++nt) {
        float bv = bias[n0b + nt * 16 + nlane];
        #pragma unroll
        for (int v = 0; v < 8; ++v) acc[nt][v] = bv;
    }

    const int nslices = K / KT;
#if HAVE_TDM
    const unsigned lds_off0 = (unsigned)(uintptr_t)(&lds_w[0][0]);
    const unsigned lds_off1 = (unsigned)(uintptr_t)(&lds_w[1][0]);
    const __bf16* wtile = Wt + (size_t)n0b * K;
#endif

    for (int i = 0; i < nslices; ++i) {
        const int buf = i & 1;
#if HAVE_TDM
        if (wave == 0) {
            if (i == 0) tdm_load_tile(wtile, lds_off0, K, Nc);
            if (i + 1 < nslices) {
                tdm_load_tile(wtile + (i + 1) * KT, (i & 1) ? lds_off0 : lds_off1, K, Nc);
                __builtin_amdgcn_s_wait_tensorcnt(1);   // current slice done; next in flight
            } else {
                __builtin_amdgcn_s_wait_tensorcnt(0);   // last slice: drain
            }
        }
#else
        for (int t = threadIdx.x; t < TN * (KT / 8); t += 256) {
            int r = t / (KT / 8), j = t % (KT / 8);
            *(v8bf*)&lds_w[buf][r * KROW + j * 8] =
                *(const v8bf*)&Wt[(size_t)(n0b + r) * K + i * KT + j * 8];
        }
#endif
        __syncthreads();
        const int k0 = i * KT;
        #pragma unroll
        for (int s = 0; s < 2; ++s) {
            // A 16x32 bf16: lanes 0-15 M=row; V0-3 K=half*8+0..7, V4-7 K=16+half*8+0..7
            v8bf alo = *(const v8bf*)(xrow + k0 + s * 32 + half * 8);
            v8bf ahi = *(const v8bf*)(xrow + k0 + s * 32 + 16 + half * 8);
            v16bf a = __builtin_shufflevector(alo, ahi, 0,1,2,3,4,5,6,7,
                                              8,9,10,11,12,13,14,15);
            // B 32x16 bf16: lane = N col; V0-3 K=half*8+0..7, V4-7 K=16+half*8+0..7
            v16bf bb[4];
            #pragma unroll
            for (int nt = 0; nt < 4; ++nt) {
                const __bf16* bp = &lds_w[buf][(nt * 16 + nlane) * KROW + s * 32 + half * 8];
                v8bf blo = *(const v8bf*)bp;
                v8bf bhi = *(const v8bf*)(bp + 16);
                bb[nt] = __builtin_shufflevector(blo, bhi, 0,1,2,3,4,5,6,7,
                                                 8,9,10,11,12,13,14,15);
            }
            #pragma unroll
            for (int nt = 0; nt < 4; ++nt)
                acc[nt] = __builtin_amdgcn_wmma_f32_16x16x32_bf16(
                              false, a, false, bb[nt], (short)0, acc[nt], false, false);
        }
        __syncthreads();
    }
    // C/D layout: lanes 0-15 N=lane, M=v; lanes 16-31 N=lane-16, M=v+8
    #pragma unroll
    for (int nt = 0; nt < 4; ++nt) {
        #pragma unroll
        for (int v = 0; v < 8; ++v) {
            int row = mtile * 16 + v + half * 8;
            if (row < M) Y[(size_t)row * Nc + n0b + nt * 16 + nlane] = acc[nt][v];
        }
    }
}

// =====================================================================
// Conversion / fill helpers
// =====================================================================
__global__ void f32_to_bf16(const float* __restrict__ in, __bf16* __restrict__ out, int n){
    int i = blockIdx.x * blockDim.x + threadIdx.x;
    if (i < n) out[i] = (__bf16)in[i];
}
// W[K][Nc] fp32 -> Wt[Nc][K] bf16 (transpose + convert; tiny)
__global__ void wt_conv(const float* __restrict__ W, __bf16* __restrict__ Wt, int K, int Nc){
    int i = blockIdx.x * blockDim.x + threadIdx.x;
    if (i >= K * Nc) return;
    int k = i / Nc, n = i - k * Nc;
    Wt[(size_t)n * K + k] = (__bf16)W[i];
}
__global__ void fill_f32(float* p, float v, int n){
    int i = blockIdx.x * blockDim.x + threadIdx.x;
    if (i < n) p[i] = v;
}
__global__ void fill_i32(int* p, int v, int n){
    int i = blockIdx.x * blockDim.x + threadIdx.x;
    if (i < n) p[i] = v;
}
__global__ void fill_bias(float* p, const float* __restrict__ bias, int n, int mod){
    int i = blockIdx.x * blockDim.x + threadIdx.x;
    if (i < n) p[i] = bias[i % mod];
}

// =====================================================================
// GATv2 edge phase
// =====================================================================
__device__ __forceinline__ void edge_sd(const int* __restrict__ ei, int e, int& s, int& d){
    if (e < N_EDGES) { s = ei[e]; d = ei[N_EDGES + e]; }
    else             { s = d = e - N_EDGES; }
}

__global__ void gat_score(const float* __restrict__ xl, const float* __restrict__ xr,
                          const float* __restrict__ att, const int* __restrict__ ei,
                          float* __restrict__ score, int* __restrict__ smax, int H)
{
    int t = blockIdx.x * blockDim.x + threadIdx.x;
    if (t >= ETOT * H) return;
    int e = t / H, h = t - e * H;
    int s, d; edge_sd(ei, e, s, d);
    int HC = H * CCH;
    const float4* pl = (const float4*)(xl + (size_t)s * HC + h * CCH);
    const float4* pr = (const float4*)(xr + (size_t)d * HC + h * CCH);
    const float4* pa = (const float4*)(att + h * CCH);
    float acc = 0.f;
    #pragma unroll
    for (int i = 0; i < 16; ++i) {
        float4 a = pl[i], b = pr[i], w = pa[i];
        float v;
        v = a.x + b.x; v = v > 0.f ? v : SLOPE_ATT * v; acc += w.x * v;
        v = a.y + b.y; v = v > 0.f ? v : SLOPE_ATT * v; acc += w.y * v;
        v = a.z + b.z; v = v > 0.f ? v : SLOPE_ATT * v; acc += w.z * v;
        v = a.w + b.w; v = v > 0.f ? v : SLOPE_ATT * v; acc += w.w * v;
    }
    score[t] = acc;
    atomicMax(&smax[d * H + h], enc_f(acc));
}

__global__ void gat_exp(float* __restrict__ score, const int* __restrict__ smax,
                        float* __restrict__ denom, const int* __restrict__ ei, int H)
{
    int t = blockIdx.x * blockDim.x + threadIdx.x;
    if (t >= ETOT * H) return;
    int e = t / H, h = t - e * H;
    int d = (e < N_EDGES) ? ei[N_EDGES + e] : (e - N_EDGES);
    float ex = __expf(score[t] - dec_f(smax[d * H + h]));
    score[t] = ex;
    atomicAdd(&denom[d * H + h], ex);
}

__global__ void gat_accum(const float* __restrict__ score, const float* __restrict__ denom,
                          const float* __restrict__ xl, const int* __restrict__ ei,
                          float* __restrict__ out, int H)
{
    int t = blockIdx.x * blockDim.x + threadIdx.x;
    if (t >= ETOT * H) return;
    int e = t / H, h = t - e * H;
    int s, d; edge_sd(ei, e, s, d);
    float alpha = score[t] / (denom[d * H + h] + 1e-16f);
    int HC = H * CCH;
    const float4* pl = (const float4*)(xl + (size_t)s * HC + h * CCH);
    float* po = out + (size_t)d * HC + h * CCH;
    #pragma unroll
    for (int i = 0; i < 16; ++i) {
        float4 a = pl[i];
        atomicAdd(po + 4*i + 0, a.x * alpha);
        atomicAdd(po + 4*i + 1, a.y * alpha);
        atomicAdd(po + 4*i + 2, a.z * alpha);
        atomicAdd(po + 4*i + 3, a.w * alpha);
    }
}

// =====================================================================
// Activations (bf16 out feeds the next GEMM's A operand)
// =====================================================================
__global__ void act_leaky_bf(const float* __restrict__ in, __bf16* __restrict__ out, int n){
    int i = blockIdx.x * blockDim.x + threadIdx.x;
    if (i >= n) return;
    float v = in[i];
    out[i] = (__bf16)(v > 0.f ? v : SLOPE_ACT * v);
}
__global__ void act_leaky_res(const float* __restrict__ in, const float* __restrict__ res,
                              float* __restrict__ out, int n){
    int i = blockIdx.x * blockDim.x + threadIdx.x;
    if (i >= n) return;
    float v = in[i];
    v = v > 0.f ? v : SLOPE_ACT * v;
    out[i] = v + res[i];
}

// =====================================================================
// Graph LayerNorm + SoftmaxAggregation
// =====================================================================
__global__ void ln_stats(const float* __restrict__ h, const int* __restrict__ batch,
                         float* __restrict__ bsum, float* __restrict__ bsq, float* __restrict__ bcnt)
{
    int n = blockIdx.x * blockDim.x + threadIdx.x;
    if (n >= N_NODES) return;
    const float4* p = (const float4*)(h + (size_t)n * CCH);
    float s = 0.f, s2 = 0.f;
    #pragma unroll
    for (int i = 0; i < 16; ++i) {
        float4 a = p[i];
        s  += a.x + a.y + a.z + a.w;
        s2 += a.x*a.x + a.y*a.y + a.z*a.z + a.w*a.w;
    }
    int b = batch[n];
    atomicAdd(&bsum[b], s);
    atomicAdd(&bsq[b],  s2);
    atomicAdd(&bcnt[b], (float)CCH);
}

__global__ void ln_finalize(const float* bsum, const float* bsq, const float* bcnt,
                            float* bmean, float* binv)
{
    int b = blockIdx.x * blockDim.x + threadIdx.x;
    if (b >= NBATCH) return;
    float cnt = fmaxf(bcnt[b], 1.0f);
    float mean = bsum[b] / cnt;
    float var  = bsq[b] / cnt - mean * mean;
    bmean[b] = mean;
    binv[b]  = rsqrtf(var + LN_EPS);
}

__global__ void ln_norm_max(const float* __restrict__ h, const int* __restrict__ batch,
                            const float* __restrict__ bmean, const float* __restrict__ binv,
                            const float* __restrict__ lnw, const float* __restrict__ lnb,
                            const float* __restrict__ tptr,
                            float* __restrict__ hn, int* __restrict__ aggm)
{
    int i = blockIdx.x * blockDim.x + threadIdx.x;
    if (i >= N_NODES * CCH) return;
    int n = i >> 6, c = i & 63;
    int b = batch[n];
    float v = (h[i] - bmean[b]) * binv[b] * lnw[c] + lnb[c];
    hn[i] = v;
    atomicMax(&aggm[b * CCH + c], enc_f(v * tptr[0]));
}

__global__ void agg_exp(const float* __restrict__ hn, const int* __restrict__ batch,
                        const float* __restrict__ tptr, const int* __restrict__ aggm,
                        float* __restrict__ exb, float* __restrict__ aggd)
{
    int i = blockIdx.x * blockDim.x + threadIdx.x;
    if (i >= N_NODES * CCH) return;
    int n = i >> 6, c = i & 63;
    int b = batch[n];
    float ex = __expf(hn[i] * tptr[0] - dec_f(aggm[b * CCH + c]));
    exb[i] = ex;
    atomicAdd(&aggd[b * CCH + c], ex);
}

__global__ void agg_final(const float* __restrict__ hn, const float* __restrict__ exb,
                          const int* __restrict__ batch, const float* __restrict__ aggd,
                          float* __restrict__ out)
{
    int i = blockIdx.x * blockDim.x + threadIdx.x;
    if (i >= N_NODES * CCH) return;
    int n = i >> 6, c = i & 63;
    int b = batch[n];
    float w = exb[i] / (aggd[b * CCH + c] + 1e-16f);
    atomicAdd(&out[b * CCH + c], w * hn[i]);
}

// =====================================================================
// Host orchestration
// =====================================================================
static inline int gblk(long n, int t){ return (int)((n + t - 1) / t); }

struct LayerP { const float *Wl, *bl, *Wr, *br, *att, *bias; };

extern "C" void kernel_launch(void* const* d_in, const int* in_sizes, int n_in,
                              void* d_out, int out_size, void* d_ws, size_t ws_size,
                              hipStream_t stream)
{
    (void)in_sizes; (void)n_in; (void)out_size; (void)ws_size;
    const float* x     = (const float*)d_in[0];   // [N, 64]
    const int*   ei    = (const int*)  d_in[1];   // [2, E] flat: src then dst
    const int*   batch = (const int*)  d_in[2];   // [N]
    int idx = 3;
    LayerP L[5];
    for (int l = 0; l < 5; ++l) {
        L[l].Wl   = (const float*)d_in[idx++];
        L[l].bl   = (const float*)d_in[idx++];
        L[l].Wr   = (const float*)d_in[idx++];
        L[l].br   = (const float*)d_in[idx++];
        L[l].att  = (const float*)d_in[idx++];
        L[l].bias = (const float*)d_in[idx++];
    }
    const float* ln_w   = (const float*)d_in[idx++];
    const float* ln_b   = (const float*)d_in[idx++];
    const float* aggr_t = (const float*)d_in[idx++];

    // ---- workspace layout (float-granular offsets) ----
    float* W = (float*)d_ws;
    size_t o = 0;
    float*  xlb   = W + o; o += (size_t)N_NODES * HC_MAX;   // lin_l output (fp32)
    float*  xrb   = W + o; o += (size_t)N_NODES * HC_MAX;   // lin_r output (fp32)
    float*  bufO  = W + o; o += (size_t)N_NODES * HC_MAX;   // GAT aggregation out
    __bf16* xbf   = (__bf16*)(W + o); o += (size_t)N_NODES * HC_MAX / 2;  // bf16 activations
    __bf16* wtb   = (__bf16*)(W + o); o += (HC_MAX * HC_MAX) / 2;         // bf16 Wt scratch
    float*  score = W + o; o += (size_t)ETOT * HEADS;
    int*    smax  = (int*)(W + o); o += (size_t)N_NODES * HEADS;
    float*  denom = W + o; o += (size_t)N_NODES * HEADS;
    float*  bsum  = W + o; o += 64;
    float*  bsq   = W + o; o += 64;
    float*  bcnt  = W + o; o += 64;
    float*  bmean = W + o; o += 64;
    float*  binv  = W + o; o += 64;
    int*    aggm  = (int*)(W + o); o += (size_t)NBATCH * CCH;
    float*  aggd  = W + o; o += (size_t)NBATCH * CCH;
    // LN-phase buffers alias xlb region (free after layer-4 accumulation)
    float* hbuf = xlb;
    float* hnb  = xlb + (size_t)N_NODES * CCH;
    float* exb  = xlb + 2 * (size_t)N_NODES * CCH;

    const int T = 256;
    // layer-0 A operand: x converted to bf16
    f32_to_bf16<<<gblk((long)N_NODES * CCH, T), T, 0, stream>>>(x, xbf, N_NODES * CCH);

    int K = CCH;   // layer 0: in_dim = 64
    for (int l = 0; l < 5; ++l) {
        int H  = (l == 4) ? 1 : HEADS;
        int HC = H * CCH;
        int Nc = HC;
        dim3 gg(gblk(N_NODES, 128), Nc / TN);
        wt_conv<<<gblk((long)K * Nc, T), T, 0, stream>>>(L[l].Wl, wtb, K, Nc);
        gemm_wmma<<<gg, 256, 0, stream>>>(xbf, wtb, L[l].bl, xlb, N_NODES, K, Nc);
        wt_conv<<<gblk((long)K * Nc, T), T, 0, stream>>>(L[l].Wr, wtb, K, Nc);
        gemm_wmma<<<gg, 256, 0, stream>>>(xbf, wtb, L[l].br, xrb, N_NODES, K, Nc);

        fill_bias<<<gblk((long)N_NODES * HC, T), T, 0, stream>>>(bufO, L[l].bias, N_NODES * HC, HC);
        fill_i32 <<<gblk(N_NODES * H, T), T, 0, stream>>>(smax, (int)0x80000000, N_NODES * H);
        fill_f32 <<<gblk(N_NODES * H, T), T, 0, stream>>>(denom, 0.f, N_NODES * H);

        int TE = ETOT * H;
        gat_score<<<gblk(TE, T), T, 0, stream>>>(xlb, xrb, L[l].att, ei, score, smax, H);
        gat_exp  <<<gblk(TE, T), T, 0, stream>>>(score, smax, denom, ei, H);
        gat_accum<<<gblk(TE, T), T, 0, stream>>>(score, denom, xlb, ei, bufO, H);

        if (l < 4) {
            act_leaky_bf<<<gblk((long)N_NODES * HC, T), T, 0, stream>>>(bufO, xbf, N_NODES * HC);
            K = HC;   // 256 for layers 1..4
        } else {
            act_leaky_res<<<gblk((long)N_NODES * CCH, T), T, 0, stream>>>(bufO, x, hbuf, N_NODES * CCH);
        }
    }

    // ---- Graph LayerNorm ----
    fill_f32<<<1, 256, 0, stream>>>(bsum, 0.f, 192);   // bsum/bsq/bcnt contiguous
    ln_stats<<<gblk(N_NODES, T), T, 0, stream>>>(hbuf, batch, bsum, bsq, bcnt);
    ln_finalize<<<1, 64, 0, stream>>>(bsum, bsq, bcnt, bmean, binv);

    // ---- Softmax aggregation ----
    fill_i32<<<gblk(NBATCH * CCH, T), T, 0, stream>>>(aggm, (int)0x80000000, NBATCH * CCH);
    fill_f32<<<gblk(NBATCH * CCH, T), T, 0, stream>>>(aggd, 0.f, NBATCH * CCH);
    ln_norm_max<<<gblk((long)N_NODES * CCH, T), T, 0, stream>>>(hbuf, batch, bmean, binv,
                                                               ln_w, ln_b, aggr_t, hnb, aggm);
    agg_exp<<<gblk((long)N_NODES * CCH, T), T, 0, stream>>>(hnb, batch, aggr_t, aggm, exb, aggd);
    fill_f32<<<gblk(NBATCH * CCH, T), T, 0, stream>>>((float*)d_out, 0.f, NBATCH * CCH);
    agg_final<<<gblk((long)N_NODES * CCH, T), T, 0, stream>>>(hnb, exb, batch, aggd, (float*)d_out);
}
